// SlidingWindowVQVAEConv_49185965474216
// MI455X (gfx1250) — matrix operand order repaired and checked by reading, not compiled, not run on hardware
//
#include <hip/hip_runtime.h>
#include <stdint.h>

// ---------------------------------------------------------------------------
// Types
// ---------------------------------------------------------------------------
typedef __attribute__((ext_vector_type(16))) __bf16 bf16x16;
typedef __attribute__((ext_vector_type(8)))  __bf16 bf16x8;
typedef __attribute__((ext_vector_type(8)))  float  f32x8;
typedef __attribute__((ext_vector_type(8)))  unsigned short u16x8;
typedef unsigned int u32x4 __attribute__((ext_vector_type(4)));
typedef unsigned int u32x8 __attribute__((ext_vector_type(8)));

#define FLAG_RELU_IN  1
#define FLAG_RELU_OUT 2
#define FLAG_UPS      4

#define USE_TDM 1   // Tensor Data Mover staging of weights into LDS

// packed bf16 relu: clear elements with sign bit set (3 packed ops)
__device__ inline bf16x8 relu_bf16x8(bf16x8 v) {
  u16x8 u = __builtin_bit_cast(u16x8, v);
  u16x8 m = u >> 15;                       // 1 where negative
  m = m * (unsigned short)0xFFFF;          // 0xFFFF where negative
  u = u & ~m;
  return __builtin_bit_cast(bf16x8, u);
}

// ---------------------------------------------------------------------------
// Support kernels
// ---------------------------------------------------------------------------
__global__ void cvt_bf16_kernel(const float* __restrict__ src,
                                __bf16* __restrict__ dst, int n) {
  int i = blockIdx.x * blockDim.x + threadIdx.x;
  if (i < n) dst[i] = (__bf16)src[i];
}

__global__ void zero_kernel(float* __restrict__ p, int n) {
  int i = blockIdx.x * blockDim.x + threadIdx.x;
  if (i < n) p[i] = 0.0f;
}

// Reorder+pad conv weight (cout,cin,k) f32 -> bf16 [coutp][k][cinp]
__global__ void wcvt_kernel(const float* __restrict__ w, __bf16* __restrict__ dst,
                            int cout, int cin, int k, int cinp, int coutp) {
  int i = blockIdx.x * blockDim.x + threadIdx.x;
  int kkp = k * cinp;
  int total = coutp * kkp;
  if (i >= total) return;
  int co = i / kkp;
  int rem = i - co * kkp;
  int kk = rem / cinp;
  int ci = rem - kk * cinp;
  float v = 0.0f;
  if (co < cout && ci < cin) v = w[((size_t)co * cin + ci) * k + kk];
  dst[i] = (__bf16)v;
}

// x (B,60,29,2) f32 -> low (B,60,32) bf16 (30 ch + pad), up (B,60,32) (28 + pad)
__global__ void split_input_kernel(const float* __restrict__ x,
                                   __bf16* __restrict__ low,
                                   __bf16* __restrict__ up, int B) {
  int i = blockIdx.x * blockDim.x + threadIdx.x;
  int nlow = B * 60 * 32;
  if (i < nlow) {
    int c = i & 31;
    int bt = i >> 5;                       // b*60+t
    float v = (c < 30) ? x[(size_t)bt * 58 + c] : 0.0f;
    low[i] = (__bf16)v;
  } else if (i < 2 * nlow) {
    int j = i - nlow;
    int c = j & 31;
    int bt = j >> 5;
    float v = (c < 28) ? x[(size_t)bt * 58 + 30 + c] : 0.0f;
    up[j] = (__bf16)v;
  }
}

// rl (B*60,30) f32, ru (B*60,28) f32 -> out (B,60,29,2) f32
__global__ void scatter_recon_kernel(const float* __restrict__ rl,
                                     const float* __restrict__ ru,
                                     float* __restrict__ out, int B) {
  int i = blockIdx.x * blockDim.x + threadIdx.x;
  int total = B * 60 * 58;
  if (i >= total) return;
  int c = i % 58;
  int bt = i / 58;
  float v;
  if (c < 30) v = rl[(size_t)bt * 30 + c];
  else        v = ru[(size_t)bt * 28 + (c - 30)];
  out[i] = v;
}

__global__ void csq_kernel(const float* __restrict__ cb,
                           float* __restrict__ csq, int K) {
  int k = blockIdx.x * blockDim.x + threadIdx.x;
  if (k < K) {
    float s = 0.0f;
    for (int d = 0; d < 128; ++d) {
      float v = cb[(size_t)k * 128 + d];
      s += v * v;
    }
    csq[k] = s;
  }
}

// ---------------------------------------------------------------------------
// TDM weight staging: DMA [Coutp][KKp] bf16 rows from global into LDS with a
// padded row stride of (KKp+8) bf16 to de-conflict the 64 LDS banks.
// Descriptor per CDNA5 ISA (iterate mode, 4 SGPR groups).
// ---------------------------------------------------------------------------
__device__ inline void tdm_stage_weights(const __bf16* w, __bf16* ldsw,
                                         int KKp, int Coutp) {
#if USE_TDM
  unsigned long long ga = (unsigned long long)(uintptr_t)w;
  unsigned lo = (unsigned)(uintptr_t)ldsw;   // LDS byte offset (low 32 bits)
  u32x4 g0;
  g0[0] = 1u;                                          // count=1, user mode
  g0[1] = lo;                                          // lds_addr
  g0[2] = (unsigned)ga;                                // global_addr[31:0]
  g0[3] = ((unsigned)(ga >> 32) & 0x01FFFFFFu) | (2u << 30);  // addr hi | type=2
  unsigned kk = (unsigned)KKp, cp = (unsigned)Coutp;
  u32x8 g1;
  g1[0] = (1u << 16) | (1u << 19);     // data_size=2B, iterate_enable
  g1[1] = (kk & 0xFFFFu) << 16;        // tensor_dim0[15:0]
  g1[2] = (kk >> 16) | (1u << 16);     // tensor_dim0[31:16] | tensor_dim1=1
  g1[3] = (kk & 0xFFFFu) << 16;        // tile_dim0 = KKp
  g1[4] = 1u;                          // tile_dim1 = 1 (tile_dim2 = 0)
  g1[5] = kk;                          // tensor_dim0_stride lo32
  g1[6] = (kk & 0xFFFFu) << 16;        // dim0_stride hi16(0) | dim1_stride lo16
  g1[7] = kk >> 16;                    // dim1_stride hi
  u32x4 g2;
  g2[0] = 1u;                          // tensor_dim2 (unused, iterate mode)
  g2[1] = (unsigned)(KKp + 8);         // lds_addr_increment (data_size units)
  g2[2] = kk;                          // global_addr_increment lo32
  g2[3] = ((cp - 1u) & 0xFFFFu) << 16; // inc hi16(0) | iterate_count = Coutp-1
  u32x4 g3;
  g3[0] = 0u; g3[1] = 0u; g3[2] = 0u; g3[3] = 0u;
  asm volatile("tensor_load_to_lds %0, %1, %2, %3"
               :: "s"(g0), "s"(g1), "s"(g2), "s"(g3) : "memory");
  __builtin_amdgcn_s_wait_tensorcnt(0);
#else
  (void)w; (void)ldsw; (void)KKp; (void)Coutp;
#endif
}

// ---------------------------------------------------------------------------
// Implicit-GEMM conv1d, WMMA bf16->f32, activations in (B,T,C) layout with
// C padded to 32. Weights pre-packed as [Coutp][K][Cinp] bf16 and staged in
// LDS (TDM) with padded row stride. Fused relu-in/relu-out/residual/upsample.
// Each of the 8 waves owns a 16-row M tile; the block shares the LDS weights.
// ---------------------------------------------------------------------------
template <int NT>
__global__ void __launch_bounds__(256)
conv1d_wmma(const __bf16* __restrict__ in, const __bf16* __restrict__ w,
            const float* __restrict__ bias, __bf16* __restrict__ outb,
            float* __restrict__ outf, const __bf16* __restrict__ res,
            int B, int Cinp, int Tin, int Cout, int Tout,
            int K, int stride, int pad, int dil, int flags) {
  extern __shared__ char smem_raw[];
  __bf16* ldsw = (__bf16*)smem_raw;

  const int KKp = Cinp * K;
  const int LDSW = KKp + 8;            // padded LDS row stride (bf16 elems)
  const int Coutp = NT * 16;
  const int M = B * Tout;
  const int lane = threadIdx.x & 31;
  const int laneN = lane & 15;
  const int half = lane >> 4;
  const int waveId = threadIdx.x >> 5;

#if USE_TDM
  if (waveId == 0) tdm_stage_weights(w, ldsw, KKp, Coutp);
#else
  for (int r = waveId; r < Coutp; r += 8)
    for (int c2 = lane * 8; c2 < KKp; c2 += 32 * 8)
      *(bf16x8*)(ldsw + (size_t)r * LDSW + c2) =
          *(const bf16x8*)(w + (size_t)r * KKp + c2);
#endif
  __syncthreads();

  const int mtile = blockIdx.x * 8 + waveId;
  const int mbase = mtile * 16;
  if (mbase >= M) return;              // wave-uniform, after barrier

  const bool relu_in  = (flags & FLAG_RELU_IN) != 0;
  const bool relu_out = (flags & FLAG_RELU_OUT) != 0;
  const bool ups      = (flags & FLAG_UPS) != 0;
  const int Teff = ups ? (2 * Tin) : Tin;
  const int perKK = Cinp >> 5;         // 32-wide chunks per kernel tap

  const int row = mbase + laneN;       // A row for this lane (both halves)
  int rb = 0, rt = 0;
  if (row < M) { rb = row / Tout; rt = row - rb * Tout; }

  f32x8 acc[NT] = {};

  for (int kk = 0; kk < K; ++kk) {
    int t = rt * stride - pad + kk * dil;
    bool okT = (row < M) && (t >= 0) && (t < Teff);
    int ts = ups ? (t >> 1) : t;
    const __bf16* ap = in + ((size_t)rb * Tin + ts) * Cinp;
    for (int cc = 0; cc < perKK; ++cc) {
      const int kbase = (kk * perKK + cc) << 5;
      bf16x8 a0 = {}, a1 = {};
      if (okT) {
        const __bf16* p = ap + (cc << 5);
        a0 = *(const bf16x8*)(p + half * 8);        // K = kbase+half*8 ..+7
        a1 = *(const bf16x8*)(p + 16 + half * 8);   // K = kbase+16+half*8 ..+7
        if (relu_in) { a0 = relu_bf16x8(a0); a1 = relu_bf16x8(a1); }
      }
      bf16x16 af;
#pragma unroll
      for (int i2 = 0; i2 < 8; ++i2) { af[i2] = a0[i2]; af[8 + i2] = a1[i2]; }
#pragma unroll
      for (int nt = 0; nt < NT; ++nt) {
        int co = nt * 16 + laneN;
        const bf16x8* bp =
            (const bf16x8*)(ldsw + (size_t)co * LDSW + kbase + half * 8);
        bf16x8 b0 = bp[0], b1 = bp[2];              // ds_load_b128 x2
        bf16x16 bfr;
#pragma unroll
        for (int i2 = 0; i2 < 8; ++i2) { bfr[i2] = b0[i2]; bfr[8 + i2] = b1[i2]; }
        acc[nt] = __builtin_amdgcn_wmma_f32_16x16x32_bf16(
            false, af, false, bfr, (short)0, acc[nt], false, false);
      }
    }
  }

  // Epilogue: bias + relu + residual; NTC output index is rr*Cout+co (no div)
#pragma unroll
  for (int nt = 0; nt < NT; ++nt) {
    int co = nt * 16 + laneN;
    if (co < Cout) {
      float bv = bias[co];
#pragma unroll
      for (int r = 0; r < 8; ++r) {
        int rr = mbase + r + half * 8;   // lanes 16-31 hold rows M+8
        if (rr < M) {
          float v = acc[nt][r] + bv;
          if (relu_out) v = fmaxf(v, 0.0f);
          size_t oidx = (size_t)rr * Cout + co;
          if (res) v += (float)res[oidx];
          if (outf) outf[oidx] = v;
          else      outb[oidx] = (__bf16)v;
        }
      }
    }
  }
}

// ---------------------------------------------------------------------------
// VQ argmin via WMMA. z rows (BT,128) bf16, codebook (512,128) bf16.
// score[m,k] = z.c_k - 0.5|c_k|^2 ; LDS u64 atomicMax argmax per row.
// ---------------------------------------------------------------------------
__device__ inline unsigned flip_f32(float f) {
  unsigned u = __float_as_uint(f);
  return (u & 0x80000000u) ? ~u : (u | 0x80000000u);
}

__global__ void __launch_bounds__(256)
vq_argmin_wmma(const __bf16* __restrict__ zb, const __bf16* __restrict__ cbb,
               const float* __restrict__ csq, int* __restrict__ idx,
               float* __restrict__ counts) {
  __shared__ unsigned long long best[16];
  const int lane = threadIdx.x & 31;
  const int laneN = lane & 15;
  const int half = lane >> 4;
  const int waveId = threadIdx.x >> 5;
  const int mbase = blockIdx.x * 16;

  if (threadIdx.x < 16) best[threadIdx.x] = 0ull;
  __syncthreads();

  const __bf16* zp = zb + (size_t)(mbase + laneN) * 128;

  f32x8 acc[4] = {};
  for (int kbase = 0; kbase < 128; kbase += 32) {
    bf16x8 a0 = *(const bf16x8*)(zp + kbase + half * 8);
    bf16x8 a1 = *(const bf16x8*)(zp + kbase + 16 + half * 8);
    bf16x16 af;
#pragma unroll
    for (int i2 = 0; i2 < 8; ++i2) { af[i2] = a0[i2]; af[8 + i2] = a1[i2]; }
#pragma unroll
    for (int nt = 0; nt < 4; ++nt) {
      int co = waveId * 64 + nt * 16 + laneN;
      const bf16x8* bp = (const bf16x8*)(cbb + (size_t)co * 128 + kbase + half * 8);
      bf16x8 b0 = bp[0], b1 = bp[2];
      bf16x16 bfr;
#pragma unroll
      for (int i2 = 0; i2 < 8; ++i2) { bfr[i2] = b0[i2]; bfr[8 + i2] = b1[i2]; }
      acc[nt] = __builtin_amdgcn_wmma_f32_16x16x32_bf16(
          false, af, false, bfr, (short)0, acc[nt], false, false);
    }
  }

#pragma unroll
  for (int nt = 0; nt < 4; ++nt) {
    int co = waveId * 64 + nt * 16 + laneN;
    float hc = 0.5f * csq[co];
#pragma unroll
    for (int r = 0; r < 8; ++r) {
      float score = acc[nt][r] - hc;
      unsigned long long key =
          ((unsigned long long)flip_f32(score) << 32) | (unsigned)co;
      atomicMax(&best[r + half * 8], key);   // ds_max_u64
    }
  }
  __syncthreads();

  if (threadIdx.x < 16) {
    int m = mbase + threadIdx.x;
    int code = (int)(best[threadIdx.x] & 0xFFFFFFFFull);
    idx[m] = code;
    atomicAdd(&counts[code], 1.0f);
  }
}

// q = codebook[idx]; sum((q-z)^2); q written bf16 (BT,128)
__global__ void __launch_bounds__(128)
gather_q_loss(const float* __restrict__ cb, const int* __restrict__ idx,
              const __bf16* __restrict__ zb, __bf16* __restrict__ qb,
              float* __restrict__ lossAcc) {
  __shared__ float red[128];
  int m = blockIdx.x;
  int d = threadIdx.x;
  int code = idx[m];
  float q = cb[(size_t)code * 128 + d];
  size_t zi = (size_t)m * 128 + d;
  float diff = q - (float)zb[zi];
  qb[zi] = (__bf16)q;
  red[d] = diff * diff;
  __syncthreads();
  for (int s = 64; s > 0; s >>= 1) {
    if (d < s) red[d] += red[d + s];
    __syncthreads();
  }
  if (d == 0) atomicAdd(lossAcc, red[0]);
}

__global__ void __launch_bounds__(512)
finalize_kernel(const float* __restrict__ countsL,
                const float* __restrict__ countsU,
                const float* __restrict__ lossAcc,
                float* __restrict__ out3, float invBT, float lossScale) {
  __shared__ float sl[512];
  __shared__ float su[512];
  int i = threadIdx.x;
  float al = countsL[i] * invBT;
  float au = countsU[i] * invBT;
  sl[i] = al * logf(al + 1e-10f);
  su[i] = au * logf(au + 1e-10f);
  __syncthreads();
  for (int s = 256; s > 0; s >>= 1) {
    if (i < s) { sl[i] += sl[i + s]; su[i] += su[i + s]; }
    __syncthreads();
  }
  if (i == 0) {
    out3[0] = (lossAcc[0] + lossAcc[1]) * lossScale;
    out3[1] = expf(-sl[0]);
    out3[2] = expf(-su[0]);
  }
}

// ---------------------------------------------------------------------------
// Host orchestration
// ---------------------------------------------------------------------------
extern "C" void kernel_launch(void* const* d_in, const int* in_sizes, int n_in,
                              void* d_out, int out_size, void* d_ws,
                              size_t ws_size, hipStream_t stream) {
  if (n_in < 135) return;
  const int B = in_sizes[0] / (60 * 58);
  const float* x = (const float*)d_in[0];

  struct Conv {
    const float* w; const float* bias;
    int cout, cin, k, cinp, coutp, kkp;
    __bf16* wbf;
  };
  int cur = 1;
  auto nf = [&]() { return (const float*)d_in[cur++]; };
  auto mkconv = [&](int cin, int cout, int k) {
    Conv c; c.w = nf(); c.bias = nf(); c.cin = cin; c.cout = cout; c.k = k;
    c.cinp = (cin + 31) & ~31;
    c.coutp = (cout > 32) ? 128 : 32;
    c.kkp = c.k * c.cinp;
    c.wbf = nullptr;
    return c;
  };
  auto walk_enc = [&](Conv* L, int cinX) {
    int n = 0;
    L[n++] = mkconv(cinX, 128, 3);
    for (int i = 0; i < 2; ++i) {
      L[n++] = mkconv(128, 128, 4);
      for (int j = 0; j < 3; ++j) {
        L[n++] = mkconv(128, 128, 3);
        L[n++] = mkconv(128, 128, 1);
      }
    }
    L[n++] = mkconv(128, 128, 3);
  };
  auto walk_dec = [&](Conv* L, int coutX) {
    int n = 0;
    L[n++] = mkconv(128, 128, 3);
    for (int i = 0; i < 2; ++i) {
      for (int j = 0; j < 3; ++j) {
        L[n++] = mkconv(128, 128, 3);
        L[n++] = mkconv(128, 128, 1);
      }
      L[n++] = mkconv(128, 128, 3);
    }
    L[n++] = mkconv(128, 128, 3);
    L[n++] = mkconv(128, coutX, 3);
  };

  Conv encL[16], encU[16], decL[17], decU[17];
  walk_enc(encL, 30);
  walk_enc(encU, 28);
  walk_dec(decL, 30);
  walk_dec(decU, 28);
  const float* cb_l = nf();
  const float* cb_u = nf();

  char* wsp = (char*)d_ws;
  size_t off = 0;
  auto alloc = [&](size_t bytes) -> void* {
    void* p = wsp + off;
    off = (off + bytes + 255) & ~(size_t)255;
    return p;
  };

  const int BT = B * 15;
  __bf16* low  = (__bf16*)alloc((size_t)B * 60 * 32 * 2);
  __bf16* up   = (__bf16*)alloc((size_t)B * 60 * 32 * 2);
  __bf16* bufA = (__bf16*)alloc((size_t)B * 60 * 128 * 2);
  __bf16* bufB = (__bf16*)alloc((size_t)B * 60 * 128 * 2);
  __bf16* bufC = (__bf16*)alloc((size_t)B * 60 * 128 * 2);
  __bf16* zl   = (__bf16*)alloc((size_t)BT * 128 * 2);
  __bf16* zu   = (__bf16*)alloc((size_t)BT * 128 * 2);
  __bf16* ql   = (__bf16*)alloc((size_t)BT * 128 * 2);
  __bf16* qu   = (__bf16*)alloc((size_t)BT * 128 * 2);
  float*  rl   = (float*)alloc((size_t)B * 60 * 30 * 4);
  float*  ru   = (float*)alloc((size_t)B * 60 * 28 * 4);
  float*  csqL = (float*)alloc(512 * 4);
  float*  csqU = (float*)alloc(512 * 4);
  float*  cntL = (float*)alloc(512 * 4);
  float*  cntU = (float*)alloc(512 * 4);
  float*  lossAcc = (float*)alloc(2 * 4);
  int*    idxL = (int*)alloc((size_t)BT * 4);
  int*    idxU = (int*)alloc((size_t)BT * 4);
  __bf16* cbLbf = (__bf16*)alloc((size_t)512 * 128 * 2);
  __bf16* cbUbf = (__bf16*)alloc((size_t)512 * 128 * 2);
  (void)ws_size;

  // weights -> padded/reordered bf16
  auto prep = [&](Conv* L, int n) {
    for (int i = 0; i < n; ++i) {
      int ne = L[i].coutp * L[i].kkp;
      L[i].wbf = (__bf16*)alloc((size_t)ne * 2);
      wcvt_kernel<<<(ne + 255) / 256, 256, 0, stream>>>(
          L[i].w, L[i].wbf, L[i].cout, L[i].cin, L[i].k, L[i].cinp, L[i].coutp);
    }
  };
  prep(encL, 16); prep(encU, 16); prep(decL, 17); prep(decU, 17);
  cvt_bf16_kernel<<<(512 * 128 + 255) / 256, 256, 0, stream>>>(cb_l, cbLbf, 512 * 128);
  cvt_bf16_kernel<<<(512 * 128 + 255) / 256, 256, 0, stream>>>(cb_u, cbUbf, 512 * 128);

  auto conv = [&](const Conv& c, const __bf16* in, __bf16* outb, float* outf,
                  const __bf16* res, int Tin, int Tout, int stride, int pad,
                  int dil, int flags) {
    int M = B * Tout;
    int blocks = ((M + 15) / 16 + 7) / 8;
    size_t sh = (size_t)c.coutp * (c.kkp + 8) * 2;
    if (c.coutp == 128)
      conv1d_wmma<8><<<blocks, 256, sh, stream>>>(
          in, c.wbf, c.bias, outb, outf, res, B, c.cinp, Tin, c.cout, Tout,
          c.k, stride, pad, dil, flags);
    else
      conv1d_wmma<2><<<blocks, 256, sh, stream>>>(
          in, c.wbf, c.bias, outb, outf, res, B, c.cinp, Tin, c.cout, Tout,
          c.k, stride, pad, dil, flags);
  };

  {
    int total = 2 * B * 60 * 32;
    split_input_kernel<<<(total + 255) / 256, 256, 0, stream>>>(x, low, up, B);
  }

  auto run_enc = [&](Conv* L, const __bf16* inp, __bf16* zout) {
    int n = 0, T = 60;
    __bf16 *c0 = bufA, *t1 = bufB, *t2 = bufC;
    conv(L[n++], inp, c0, nullptr, nullptr, T, T, 1, 1, 1, FLAG_RELU_OUT);
    const int dils[3] = {1, 2, 4};
    for (int i = 0; i < 2; ++i) {
      int Tn = T / 2;
      conv(L[n++], c0, t1, nullptr, nullptr, T, Tn, 2, 1, 1, 0);
      { __bf16* tmp = c0; c0 = t1; t1 = tmp; }
      T = Tn;
      for (int j = 0; j < 3; ++j) {
        int dl = dils[j];
        conv(L[n++], c0, t1, nullptr, nullptr, T, T, 1, dl, dl, FLAG_RELU_IN);
        conv(L[n++], t1, t2, nullptr, c0, T, T, 1, 0, 1, FLAG_RELU_IN);
        { __bf16* tmp = c0; c0 = t2; t2 = tmp; }
      }
    }
    conv(L[n++], c0, zout, nullptr, nullptr, T, T, 1, 1, 1, 0);
  };
  run_enc(encL, low, zl);
  run_enc(encU, up, zu);

  zero_kernel<<<2, 256, 0, stream>>>(cntL, 512);
  zero_kernel<<<2, 256, 0, stream>>>(cntU, 512);
  zero_kernel<<<1, 32, 0, stream>>>(lossAcc, 2);
  csq_kernel<<<8, 64, 0, stream>>>(cb_l, csqL, 512);
  csq_kernel<<<8, 64, 0, stream>>>(cb_u, csqU, 512);
  vq_argmin_wmma<<<BT / 16, 256, 0, stream>>>(zl, cbLbf, csqL, idxL, cntL);
  vq_argmin_wmma<<<BT / 16, 256, 0, stream>>>(zu, cbUbf, csqU, idxU, cntU);
  gather_q_loss<<<BT, 128, 0, stream>>>(cb_l, idxL, zl, ql, &lossAcc[0]);
  gather_q_loss<<<BT, 128, 0, stream>>>(cb_u, idxU, zu, qu, &lossAcc[1]);

  auto run_dec = [&](Conv* L, const __bf16* q, float* rout) {
    int n = 0, T = 15;
    __bf16 *c0 = bufA, *t1 = bufB, *t2 = bufC;
    conv(L[n++], q, c0, nullptr, nullptr, T, T, 1, 1, 1, FLAG_RELU_OUT);
    const int dils[3] = {4, 2, 1};
    for (int i = 0; i < 2; ++i) {
      for (int j = 0; j < 3; ++j) {
        int dl = dils[j];
        conv(L[n++], c0, t1, nullptr, nullptr, T, T, 1, dl, dl, FLAG_RELU_IN);
        conv(L[n++], t1, t2, nullptr, c0, T, T, 1, 0, 1, FLAG_RELU_IN);
        { __bf16* tmp = c0; c0 = t2; t2 = tmp; }
      }
      conv(L[n++], c0, t1, nullptr, nullptr, T, 2 * T, 1, 1, 1, FLAG_UPS);
      { __bf16* tmp = c0; c0 = t1; t1 = tmp; }
      T *= 2;
    }
    conv(L[n++], c0, t1, nullptr, nullptr, T, T, 1, 1, 1, FLAG_RELU_OUT);
    { __bf16* tmp = c0; c0 = t1; t1 = tmp; }
    conv(L[n++], c0, nullptr, rout, nullptr, T, T, 1, 1, 1, 0);
  };
  run_dec(decL, ql, rl);
  run_dec(decU, qu, ru);

  int reconN = B * 60 * 58;
  scatter_recon_kernel<<<(reconN + 255) / 256, 256, 0, stream>>>(
      rl, ru, (float*)d_out, B);
  float invBT = 1.0f / (float)BT;
  float lossScale = 1.25f / ((float)BT * 128.0f);
  finalize_kernel<<<1, 512, 0, stream>>>(cntL, cntU, lossAcc,
                                         (float*)d_out + reconN, invBT,
                                         lossScale);
  (void)out_size;
}
